// MultiHeadSelfAttention_26293789786283
// MI455X (gfx1250) — matrix operand order repaired and checked by reading, not compiled, not run on hardware
//
#include <hip/hip_runtime.h>

// ---------------------------------------------------------------------------
// MI455X (gfx1250) multi-head self-attention, bf16 WMMA path.
// All matmuls use v_wmma_f32_16x16x32_bf16 (f32 accumulate), softmax in f32.
// ---------------------------------------------------------------------------

typedef __bf16 bf16_t;
typedef __attribute__((ext_vector_type(16))) __bf16 v16bf;
typedef __attribute__((ext_vector_type(8)))  __bf16 v8bf;
typedef __attribute__((ext_vector_type(8)))  float  v8f;

#define D_MODEL 1024
#define N_HEADS 16
#define SEQ     2048
#define BATCH   2
#define DK      64   // head dim

// round-to-nearest-even f32 -> bf16
__device__ __forceinline__ bf16_t f2bf(float f) {
  unsigned u = __builtin_bit_cast(unsigned, f);
  u += 0x7FFFu + ((u >> 16) & 1u);
  unsigned short h = (unsigned short)(u >> 16);
  return __builtin_bit_cast(bf16_t, h);
}

// A-fragment (16-bit A, 16x32 MxK): lane = M (mod 16); per-lane K pattern:
//   elems 0..7  -> K = hl*8 + i       (hl = lane/16)
//   elems 8..15 -> K = 16 + hl*8 + i
__device__ __forceinline__ v16bf load_a(const bf16_t* row_k0, int hl) {
  v8bf lo = *(const v8bf*)(row_k0 + hl * 8);
  v8bf hi = *(const v8bf*)(row_k0 + 16 + hl * 8);
  v16bf r;
#pragma unroll
  for (int i = 0; i < 8; ++i) { r[i] = lo[i]; r[i + 8] = hi[i]; }
  return r;
}

// B-fragment (16-bit B, 32x16 KxN): lane = N (mod 16); same K pattern as A:
//   elems 0..7 -> K = hl*8 + i, elems 8..15 -> K = 16 + hl*8 + i.
// `col` points at the 32 contiguous K-values of this lane's column.
__device__ __forceinline__ v16bf load_b(const bf16_t* col, int hl) {
  v8bf lo = *(const v8bf*)(col + hl * 8);
  v8bf hi = *(const v8bf*)(col + 16 + hl * 8);
  v16bf r;
#pragma unroll
  for (int i = 0; i < 8; ++i) { r[i] = lo[i]; r[i + 8] = hi[i]; }
  return r;
}

__device__ __forceinline__ v8f wmma_bf16(v16bf a, v16bf b, v8f c) {
  return __builtin_amdgcn_wmma_f32_16x16x32_bf16(
      /*neg_a=*/false, a, /*neg_b=*/false, b,
      /*c_mod=*/(short)0, c, /*reuse_a=*/false, /*reuse_b=*/false);
}

__global__ void cvt_f32_bf16(const float* __restrict__ src,
                             bf16_t* __restrict__ dst, int n) {
  int i = blockIdx.x * blockDim.x + threadIdx.x;
  if (i < n) dst[i] = f2bf(src[i]);
}

// ---------------------------------------------------------------------------
// C[4096 x 1024] = A[4096 x 1024] @ W[1024 x 1024]^T + bias
// mode 0: f32 row-major to outF
// mode 1: bf16 scatter to head layout (bh, S, dk)   (for qh / kh)
// mode 2: bf16 scatter to transposed  (bh, dk, S)   (for vt)
// Block = 128 threads (4 waves); each wave: 16 rows x 64 cols; WG tile 64x64.
// ---------------------------------------------------------------------------
__global__ __launch_bounds__(128) void gemm_bf16(
    const bf16_t* __restrict__ A, const bf16_t* __restrict__ W,
    const float* __restrict__ bias,
    float* __restrict__ outF, bf16_t* __restrict__ outB, int mode) {
  const int K = D_MODEL;
  const int wave = threadIdx.x >> 5;
  const int lane = threadIdx.x & 31;
  const int hl   = lane >> 4;
  const int l16  = lane & 15;
  const int mbase = blockIdx.x * 64 + wave * 16;
  const int nbase = blockIdx.y * 64;

  v8f acc[4] = {};
  const bf16_t* arow = A + (size_t)(mbase + l16) * K;

  for (int k0 = 0; k0 < K; k0 += 32) {
    v16bf a = load_a(arow + k0, hl);
#pragma unroll
    for (int nt = 0; nt < 4; ++nt) {
      const bf16_t* wcol = W + (size_t)(nbase + nt * 16 + l16) * K + k0;
      acc[nt] = wmma_bf16(a, load_b(wcol, hl), acc[nt]);
    }
  }

  // C layout: lane holds column n = nbase+nt*16+l16; VGPR r = row r + 8*hl.
#pragma unroll
  for (int nt = 0; nt < 4; ++nt) {
    const int n  = nbase + nt * 16 + l16;
    const float bv = bias[n];
#pragma unroll
    for (int r = 0; r < 8; ++r) {
      const int row = mbase + r + hl * 8;   // row = s*BATCH + b
      const float val = acc[nt][r] + bv;
      if (mode == 0) {
        outF[(size_t)row * D_MODEL + n] = val;
      } else {
        const int s = row >> 1, b = row & 1;
        const int h = n >> 6,  d = n & 63;
        size_t idx;
        if (mode == 1) idx = ((size_t)(b * N_HEADS + h) * SEQ + s) * DK + d;
        else           idx = ((size_t)(b * N_HEADS + h) * DK + d) * SEQ + s;
        outB[idx] = f2bf(val);
      }
    }
  }
}

// ---------------------------------------------------------------------------
// Flash attention: one block per (64-query tile, head). 4 waves; each wave
// owns 16 query rows. Online softmax; probs transposed via per-wave LDS tile.
//   Qh,Kh: (32, 2048, 64) bf16   Vt: (32, 64, 2048) bf16
//   Out:   (4096, 1024) bf16, row = s*BATCH + b, col = h*64 + d
// ---------------------------------------------------------------------------
__global__ __launch_bounds__(128) void flash_attn(
    const bf16_t* __restrict__ Qh, const bf16_t* __restrict__ Kh,
    const bf16_t* __restrict__ Vt, bf16_t* __restrict__ Out) {
  __shared__ __align__(16) bf16_t Plds[4][16][32];

  const int wave = threadIdx.x >> 5;
  const int lane = threadIdx.x & 31;
  const int hl   = lane >> 4;
  const int l16  = lane & 15;
  const int bh   = blockIdx.y;                 // b*16 + h
  const int qbase = blockIdx.x * 64 + wave * 16;

  const bf16_t* Qb = Qh + (size_t)bh * SEQ * DK;
  const bf16_t* Kb = Kh + (size_t)bh * SEQ * DK;
  const bf16_t* Vb = Vt + (size_t)bh * DK * SEQ;

  // Q A-fragments for this wave's 16 rows, dk split 0..31 / 32..63
  const bf16_t* qrow = Qb + (size_t)(qbase + l16) * DK;
  const v16bf qa0 = load_a(qrow, hl);
  const v16bf qa1 = load_a(qrow + 32, hl);

  v8f o[4] = {};
  float mrow[8], lrow[8];
#pragma unroll
  for (int r = 0; r < 8; ++r) { mrow[r] = -1e30f; lrow[r] = 0.0f; }

  for (int kb = 0; kb < SEQ; kb += 32) {
    // ---- scores: two 16-key fragments, K dim = dk = 64 (2 WMMAs each) ----
    v8f s0 = {}, s1 = {};
    {
      const bf16_t* kcol0 = Kb + (size_t)(kb + l16) * DK;
      s0 = wmma_bf16(qa0, load_b(kcol0, hl), s0);
      s0 = wmma_bf16(qa1, load_b(kcol0 + 32, hl), s0);
      const bf16_t* kcol1 = Kb + (size_t)(kb + 16 + l16) * DK;
      s1 = wmma_bf16(qa0, load_b(kcol1, hl), s1);
      s1 = wmma_bf16(qa1, load_b(kcol1 + 32, hl), s1);
    }
    // ---- online softmax (rows split: VGPR r <-> row r + 8*hl; 16 lanes/row)
    float corr[8];
#pragma unroll
    for (int r = 0; r < 8; ++r) {
      s0[r] *= 0.125f;  // 1/sqrt(dk)
      s1[r] *= 0.125f;
      float v = fmaxf(s0[r], s1[r]);
      v = fmaxf(v, __shfl_xor(v, 1, 32));
      v = fmaxf(v, __shfl_xor(v, 2, 32));
      v = fmaxf(v, __shfl_xor(v, 4, 32));
      v = fmaxf(v, __shfl_xor(v, 8, 32));
      const float nm = fmaxf(mrow[r], v);
      corr[r] = __expf(mrow[r] - nm);
      mrow[r] = nm;
      const float p0 = __expf(s0[r] - nm);
      const float p1 = __expf(s1[r] - nm);
      s0[r] = p0; s1[r] = p1;
      float t = p0 + p1;
      t += __shfl_xor(t, 1, 32);
      t += __shfl_xor(t, 2, 32);
      t += __shfl_xor(t, 4, 32);
      t += __shfl_xor(t, 8, 32);
      lrow[r] = lrow[r] * corr[r] + t;
    }
#pragma unroll
    for (int nt = 0; nt < 4; ++nt)
#pragma unroll
      for (int r = 0; r < 8; ++r) o[nt][r] *= corr[r];

    // ---- transpose P through this wave's private LDS tile ----
#pragma unroll
    for (int r = 0; r < 8; ++r) {
      Plds[wave][r + hl * 8][l16]      = f2bf(s0[r]);
      Plds[wave][r + hl * 8][l16 + 16] = f2bf(s1[r]);
    }
    // same-wave LDS RAW: DS ops in-order; compiler inserts s_wait_dscnt
    const v16bf pf = load_a(&Plds[wave][l16][0], hl);

    // ---- O += P(16x32) @ V(32x64): 4 n-tiles ----
#pragma unroll
    for (int nt = 0; nt < 4; ++nt) {
      const bf16_t* vcol = Vb + (size_t)(nt * 16 + l16) * SEQ + kb;
      o[nt] = wmma_bf16(pf, load_b(vcol, hl), o[nt]);
    }
  }

  // ---- finalize: divide by row sums, scatter to (S*B, D) bf16 ----
  const int b = bh >> 4, h = bh & 15;
#pragma unroll
  for (int nt = 0; nt < 4; ++nt) {
    const int d = nt * 16 + l16;
#pragma unroll
    for (int r = 0; r < 8; ++r) {
      const int qi = qbase + r + hl * 8;
      const float val = o[nt][r] / lrow[r];
      Out[(size_t)(qi * BATCH + b) * D_MODEL + h * DK + d] = f2bf(val);
    }
  }
}

// ---------------------------------------------------------------------------
extern "C" void kernel_launch(void* const* d_in, const int* in_sizes, int n_in,
                              void* d_out, int out_size, void* d_ws, size_t ws_size,
                              hipStream_t stream) {
  const float* q    = (const float*)d_in[0];
  const float* k    = (const float*)d_in[1];
  const float* v    = (const float*)d_in[2];
  const float* wIn  = (const float*)d_in[3];  // (3072, 1024)
  const float* bIn  = (const float*)d_in[4];  // (3072,)
  const float* wOut = (const float*)d_in[5];  // (1024, 1024)
  const float* bOut = (const float*)d_in[6];  // (1024,)

  const size_t NTOK = (size_t)SEQ * BATCH;            // 4096
  const size_t MK   = NTOK * D_MODEL;                 // 4.19M elems
  const size_t WSZ  = (size_t)D_MODEL * D_MODEL;      // 1.05M elems

  char* base = (char*)d_ws;
  size_t off = 0;
  auto take = [&](size_t elems) -> bf16_t* {
    bf16_t* p = (bf16_t*)(base + off);
    off = (off + elems * sizeof(bf16_t) + 255) & ~(size_t)255;
    return p;
  };
  bf16_t* qbf    = take(MK);
  bf16_t* kbf    = take(MK);
  bf16_t* vbf    = take(MK);
  bf16_t* wInBf  = take(3 * WSZ);
  bf16_t* wOutBf = take(WSZ);
  bf16_t* qh     = take(MK);
  bf16_t* kh     = take(MK);
  bf16_t* vt     = take(MK);
  bf16_t* attnb  = qbf;  // qbf is dead after the Q projection; reuse for attn out

  // 1) f32 -> bf16 converts (streaming, ~100 MB total @ 23.3 TB/s: negligible)
  cvt_f32_bf16<<<dim3((unsigned)((MK + 255) / 256)), dim3(256), 0, stream>>>(q, qbf, (int)MK);
  cvt_f32_bf16<<<dim3((unsigned)((MK + 255) / 256)), dim3(256), 0, stream>>>(k, kbf, (int)MK);
  cvt_f32_bf16<<<dim3((unsigned)((MK + 255) / 256)), dim3(256), 0, stream>>>(v, vbf, (int)MK);
  cvt_f32_bf16<<<dim3((unsigned)((3 * WSZ + 255) / 256)), dim3(256), 0, stream>>>(wIn, wInBf, (int)(3 * WSZ));
  cvt_f32_bf16<<<dim3((unsigned)((WSZ + 255) / 256)), dim3(256), 0, stream>>>(wOut, wOutBf, (int)WSZ);

  // 2) QKV projections (WMMA GEMMs), scattering into head layouts
  dim3 gblk(128), ggrid((unsigned)(NTOK / 64), D_MODEL / 64);
  gemm_bf16<<<ggrid, gblk, 0, stream>>>(qbf, wInBf,           bIn,               nullptr, qh, 1);
  gemm_bf16<<<ggrid, gblk, 0, stream>>>(kbf, wInBf + WSZ,     bIn + D_MODEL,     nullptr, kh, 1);
  gemm_bf16<<<ggrid, gblk, 0, stream>>>(vbf, wInBf + 2 * WSZ, bIn + 2 * D_MODEL, nullptr, vt, 2);

  // 3) flash attention: grid = (query tiles, B*H heads)
  flash_attn<<<dim3(SEQ / 64, BATCH * N_HEADS), gblk, 0, stream>>>(qh, kh, vt, attnb);

  // 4) output projection -> f32 d_out
  gemm_bf16<<<ggrid, gblk, 0, stream>>>(attnb, wOutBf, bOut, (float*)d_out, nullptr, 0);
}
